// VolatilityAwareMultiHeadAttention_53017076302098
// MI455X (gfx1250) — matrix operand order repaired and checked by reading, not compiled, not run on hardware
//
#include <hip/hip_runtime.h>
#include <hip/hip_bf16.h>

// Problem constants (match reference)
#define BATCH  256
#define SEQ    128
#define DMODEL 1024
#define NHEADS 8
#define DEPTH  128

typedef __bf16 bf16_t;
typedef __attribute__((ext_vector_type(16))) __bf16 v16bf;
typedef __attribute__((ext_vector_type(8)))  __bf16 v8bf;
typedef __attribute__((ext_vector_type(8)))  float  v8f;

union FragBF { v16bf v; v8bf h[2]; };

__device__ __forceinline__ v8f vzero8() {
  v8f z;
#pragma unroll
  for (int i = 0; i < 8; ++i) z[i] = 0.0f;
  return z;
}

// Per-lane async copy of 16 bytes global -> LDS (CDNA5 GLOBAL_LOAD_ASYNC_TO_LDS_B128,
// tracked by ASYNCcnt).  lds_off = 32-bit LDS byte offset (flat LDS addr truncates
// to the LDS offset per ISA 10.2), gaddr = per-lane global address.
__device__ __forceinline__ void async_copy_b128(unsigned int lds_off,
                                                const void* gaddr) {
  asm volatile("global_load_async_to_lds_b128 %0, %1, off"
               :
               : "v"(lds_off), "v"(gaddr)
               : "memory");
}

__device__ __forceinline__ void wait_asynccnt0() {
  asm volatile("s_wait_asynccnt 0x0" ::: "memory");
}

// A fragment (16x32 bf16, row-major in LDS, ld = row stride in elements).
// Lane<16: row=lane, K 0..7 (vgpr0-3) and 16..23 (vgpr4-7);
// Lane>=16: row=lane-16, K 8..15 and 24..31.  (ISA 7.12.2 16-bit A table)
__device__ __forceinline__ v16bf load_frag_a(const bf16_t* base, int ld) {
  const int lane = threadIdx.x & 31;
  const int row  = lane & 15;
  const int koff = (lane >> 4) << 3;   // 0 or 8
  FragBF f;
  f.h[0] = *(const v8bf*)(base + row * ld + koff);
  f.h[1] = *(const v8bf*)(base + row * ld + 16 + koff);
  return f.v;
}

// B fragment (32x16 bf16) loaded from B^T stored row-major [N][K] in LDS.
// Lane<16: col=lane, K 0..7 and 16..23; lane>=16: col=lane-16, K 8..15 and 24..31.
__device__ __forceinline__ v16bf load_frag_bT(const bf16_t* baseT, int ld) {
  const int lane = threadIdx.x & 31;
  const int col  = lane & 15;
  const int koff = (lane >> 4) << 3;   // 0 or 8
  FragBF f;
  f.h[0] = *(const v8bf*)(baseT + col * ld + koff);
  f.h[1] = *(const v8bf*)(baseT + col * ld + 16 + koff);
  return f.v;
}

__device__ __forceinline__ v8f wmma_bf16(v16bf a, v16bf b, v8f c) {
  return __builtin_amdgcn_wmma_f32_16x16x32_bf16(
      /*neg_a=*/false, a, /*neg_b=*/false, b,
      /*c_mod=*/(short)0, c, /*reuse_a=*/false, /*reuse_b=*/false);
}

// ---------------------------------------------------------------------------
// Projection GEMM: out[b,h,s,dd] = (X @ W + bias), X:[B*S, D] f32, W:[D,D] f32
// Tile 128x128 per workgroup, K-step 32, 8 waves (one 16-row strip each).
// OUT_MODE 0 -> bf16 head-split output; OUT_MODE 1 -> f32 head-split output.
// ---------------------------------------------------------------------------
template <int OUT_MODE>
__global__ __launch_bounds__(256) void proj_kernel(
    const float* __restrict__ X, const float* __restrict__ W,
    const float* __restrict__ bias, void* __restrict__ out) {
  __shared__ bf16_t a_lds[128 * 40];   // [m][k] row-major, pad to 40
  __shared__ bf16_t wT_lds[128 * 40];  // [n][k] (W transposed), pad to 40

  const int t    = threadIdx.x;
  const int wave = t >> 5;
  const int n0   = blockIdx.x * 128;
  const int m0   = blockIdx.y * 128;

  v8f acc[8];
#pragma unroll
  for (int i = 0; i < 8; ++i) acc[i] = vzero8();

  for (int k0 = 0; k0 < DMODEL; k0 += 32) {
    __syncthreads();
    // A tile: 128 rows x 32 K, fp32 -> bf16
#pragma unroll
    for (int i = 0; i < 16; ++i) {
      const int idx = i * 256 + t;
      const int m = idx >> 5, k = idx & 31;
      a_lds[m * 40 + k] = (bf16_t)X[(size_t)(m0 + m) * DMODEL + k0 + k];
    }
    // W tile transposed: wT[n][k] = W[k0+k][n0+n]
#pragma unroll
    for (int i = 0; i < 16; ++i) {
      const int idx = i * 256 + t;
      const int k = idx >> 7, n = idx & 127;
      wT_lds[n * 40 + k] = (bf16_t)W[(size_t)(k0 + k) * DMODEL + n0 + n];
    }
    __syncthreads();

    if (k0 + 32 < DMODEL) {  // speculative prefetch of next tiles
      __builtin_prefetch(&X[(size_t)(m0 + (t >> 1)) * DMODEL + k0 + 32], 0, 0);
      __builtin_prefetch(&W[(size_t)(k0 + 32 + (t & 31)) * DMODEL + n0], 0, 0);
    }

    const v16bf af = load_frag_a(a_lds + wave * 16 * 40, 40);
#pragma unroll
    for (int ct = 0; ct < 8; ++ct) {
      const v16bf bfb = load_frag_bT(wT_lds + ct * 16 * 40, 40);
      acc[ct] = wmma_bf16(af, bfb, acc[ct]);
    }
  }

  // Epilogue: bias add + head-split store
  const int lane   = t & 31;
  const int colb   = lane & 15;
  const int rowoff = (lane >> 4) << 3;
#pragma unroll
  for (int ct = 0; ct < 8; ++ct) {
#pragma unroll
    for (int r = 0; r < 8; ++r) {
      const int m = m0 + wave * 16 + r + rowoff;
      const int n = n0 + ct * 16 + colb;
      const float val = acc[ct][r] + bias[n];
      const int b = m >> 7, s = m & 127;
      const int h = n >> 7, dd = n & 127;
      const size_t o = (((size_t)b * NHEADS + h) * SEQ + s) * DEPTH + dd;
      if (OUT_MODE == 0)
        ((bf16_t*)out)[o] = (bf16_t)val;
      else
        ((float*)out)[o] = val;
    }
  }
}

// ---------------------------------------------------------------------------
// Fused per-head attention: one workgroup per (b,h).
// logits = (Q K^T)/sqrt(128) + mask*(-1e9);  vol = logits*(V+noise);
// attn = softmax(vol) -> d_out (f32) and bf16 -> P;  out_head = P @ V (bf16).
// Q/K tiles staged with CDNA5 async global->LDS copies (ASYNCcnt).
// ---------------------------------------------------------------------------
__global__ __launch_bounds__(256) void attn_kernel(
    const bf16_t* __restrict__ Q, const bf16_t* __restrict__ K,
    const float* __restrict__ V, const float* __restrict__ noise,
    const float* __restrict__ mask, float* __restrict__ attn_out,
    bf16_t* __restrict__ out_head) {
  extern __shared__ __align__(16) char smem[];
  bf16_t* q_lds  = (bf16_t*)smem;          // [128][136]
  bf16_t* k_lds  = q_lds + 128 * 136;      // [128][136]
  bf16_t* vT_lds = k_lds + 128 * 136;      // [d][kk] = V^T, [128][136]
  bf16_t* p_lds  = vT_lds + 128 * 136;     // attn probs bf16 [128][136]

  const int t    = threadIdx.x;
  const int wave = t >> 5;
  const int lane = t & 31;
  const size_t bh = blockIdx.x;

  const bf16_t* q  = Q + bh * SEQ * DEPTH;
  const bf16_t* k  = K + bh * SEQ * DEPTH;
  const float*  v  = V + bh * SEQ * DEPTH;
  const float*  nz = noise + bh * SEQ * DEPTH;
  float* pout = attn_out + bh * SEQ * SEQ;
  bf16_t* oh  = out_head + bh * SEQ * DEPTH;

  // Stage Q/K with async global->LDS b128 copies (no VGPR round-trip; the
  // copies run in the background while we do the V^T convert/transpose below).
#pragma unroll
  for (int i = 0; i < 8; ++i) {
    const int vecid = i * 256 + t;          // 2048 v8bf chunks per matrix
    const int row = vecid >> 4;
    const int c8  = (vecid & 15) << 3;
    async_copy_b128((unsigned int)(uintptr_t)(q_lds + row * 136 + c8),
                    (const void*)(q + row * 128 + c8));
    async_copy_b128((unsigned int)(uintptr_t)(k_lds + row * 136 + c8),
                    (const void*)(k + row * 128 + c8));
  }
  // V^T staging (needs fp32 -> bf16 conversion + transpose, so done manually).
#pragma unroll
  for (int i = 0; i < 64; ++i) {
    const int idx = i * 256 + t;            // 16384 elements
    const int kk = idx >> 7, d = idx & 127;
    vT_lds[d * 136 + kk] = (bf16_t)v[idx];
  }
  wait_asynccnt0();   // own wave's async fills complete
  __syncthreads();    // ... and everyone else's

  // ---- QK^T: wave w owns query rows 16w..16w+15, all 8 col tiles ----
  v8f acc[8];
#pragma unroll
  for (int i = 0; i < 8; ++i) acc[i] = vzero8();

  const bf16_t* qa = q_lds + wave * 16 * 136;
#pragma unroll
  for (int ks = 0; ks < 4; ++ks) {
    const v16bf af = load_frag_a(qa + ks * 32, 136);
#pragma unroll
    for (int ct = 0; ct < 8; ++ct) {
      const v16bf bfb = load_frag_bT(k_lds + ct * 16 * 136 + ks * 32, 136);
      acc[ct] = wmma_bf16(af, bfb, acc[ct]);
    }
  }

  // ---- mask, volatility scale, register softmax (16-lane shfl reduce) ----
  const float scale = 0.08838834764831845f;  // 1/sqrt(128)
  const int colb   = lane & 15;
  const int rowoff = (lane >> 4) << 3;
#pragma unroll
  for (int r = 0; r < 8; ++r) {
    const int i = wave * 16 + r + rowoff;
    float rowmax = -3.0e38f;
#pragma unroll
    for (int ct = 0; ct < 8; ++ct) {
      const int j = ct * 16 + colb;
      float lg = acc[ct][r] * scale + mask[i * SEQ + j] * -1.0e9f;
      const float vg = v[i * DEPTH + j] + nz[i * DEPTH + j];
      lg *= vg;                      // vol_logits = logits * v_garch
      acc[ct][r] = lg;
      rowmax = fmaxf(rowmax, lg);
    }
    rowmax = fmaxf(rowmax, __shfl_xor(rowmax, 1, 32));
    rowmax = fmaxf(rowmax, __shfl_xor(rowmax, 2, 32));
    rowmax = fmaxf(rowmax, __shfl_xor(rowmax, 4, 32));
    rowmax = fmaxf(rowmax, __shfl_xor(rowmax, 8, 32));
    float rsum = 0.0f;
#pragma unroll
    for (int ct = 0; ct < 8; ++ct) {
      const float e = __expf(acc[ct][r] - rowmax);
      acc[ct][r] = e;
      rsum += e;
    }
    rsum += __shfl_xor(rsum, 1, 32);
    rsum += __shfl_xor(rsum, 2, 32);
    rsum += __shfl_xor(rsum, 4, 32);
    rsum += __shfl_xor(rsum, 8, 32);
    const float inv = 1.0f / rsum;
#pragma unroll
    for (int ct = 0; ct < 8; ++ct) {
      const int j = ct * 16 + colb;
      const float pp = acc[ct][r] * inv;
      pout[(size_t)i * SEQ + j] = pp;         // attn output (f32)
      p_lds[i * 136 + j] = (bf16_t)pp;        // bf16 operand for PV
    }
  }
  __syncthreads();

  // ---- P @ V: A = P rows, B^T = V^T in LDS ----
  v8f oacc[8];
#pragma unroll
  for (int i = 0; i < 8; ++i) oacc[i] = vzero8();

  const bf16_t* pa = p_lds + wave * 16 * 136;
#pragma unroll
  for (int ks = 0; ks < 4; ++ks) {
    const v16bf af = load_frag_a(pa + ks * 32, 136);
#pragma unroll
    for (int ct = 0; ct < 8; ++ct) {
      const v16bf bfb = load_frag_bT(vT_lds + ct * 16 * 136 + ks * 32, 136);
      oacc[ct] = wmma_bf16(af, bfb, oacc[ct]);
    }
  }
#pragma unroll
  for (int ct = 0; ct < 8; ++ct) {
#pragma unroll
    for (int r = 0; r < 8; ++r) {
      const int i = wave * 16 + r + rowoff;
      const int d = ct * 16 + colb;
      oh[i * DEPTH + d] = (bf16_t)oacc[ct][r];
    }
  }
}

// ---------------------------------------------------------------------------
// Output projection: A = out_head (bf16, [B,H,S,DEPTH] layout, merged on load),
// out[b*S+s, :] = A @ wo + bo  (f32, written straight into d_out).
// ---------------------------------------------------------------------------
__global__ __launch_bounds__(256) void final_proj_kernel(
    const bf16_t* __restrict__ Ah, const float* __restrict__ W,
    const float* __restrict__ bias, float* __restrict__ out) {
  __shared__ bf16_t a_lds[128 * 40];
  __shared__ bf16_t wT_lds[128 * 40];

  const int t    = threadIdx.x;
  const int wave = t >> 5;
  const int n0   = blockIdx.x * 128;
  const int m0   = blockIdx.y * 128;

  v8f acc[8];
#pragma unroll
  for (int i = 0; i < 8; ++i) acc[i] = vzero8();

  for (int k0 = 0; k0 < DMODEL; k0 += 32) {
    __syncthreads();
#pragma unroll
    for (int i = 0; i < 16; ++i) {
      const int idx = i * 256 + t;
      const int m = idx >> 5, kloc = idx & 31;
      const int mg = m0 + m;
      const int b = mg >> 7, s = mg & 127;
      const int kg = k0 + kloc;
      const int h = kg >> 7, dd = kg & 127;
      a_lds[m * 40 + kloc] =
          Ah[(((size_t)b * NHEADS + h) * SEQ + s) * DEPTH + dd];
    }
#pragma unroll
    for (int i = 0; i < 16; ++i) {
      const int idx = i * 256 + t;
      const int kloc = idx >> 7, n = idx & 127;
      wT_lds[n * 40 + kloc] = (bf16_t)W[(size_t)(k0 + kloc) * DMODEL + n0 + n];
    }
    __syncthreads();

    if (k0 + 32 < DMODEL)
      __builtin_prefetch(&W[(size_t)(k0 + 32 + (t & 31)) * DMODEL + n0], 0, 0);

    const v16bf af = load_frag_a(a_lds + wave * 16 * 40, 40);
#pragma unroll
    for (int ct = 0; ct < 8; ++ct) {
      const v16bf bfb = load_frag_bT(wT_lds + ct * 16 * 40, 40);
      acc[ct] = wmma_bf16(af, bfb, acc[ct]);
    }
  }

  const int lane   = t & 31;
  const int colb   = lane & 15;
  const int rowoff = (lane >> 4) << 3;
#pragma unroll
  for (int ct = 0; ct < 8; ++ct) {
#pragma unroll
    for (int r = 0; r < 8; ++r) {
      const int m = m0 + wave * 16 + r + rowoff;
      const int n = n0 + ct * 16 + colb;
      out[(size_t)m * DMODEL + n] = acc[ct][r] + bias[n];
    }
  }
}

// ---------------------------------------------------------------------------
extern "C" void kernel_launch(void* const* d_in, const int* in_sizes, int n_in,
                              void* d_out, int out_size, void* d_ws,
                              size_t ws_size, hipStream_t stream) {
  (void)in_sizes; (void)n_in; (void)out_size; (void)ws_size;
  const float* q     = (const float*)d_in[0];
  const float* k     = (const float*)d_in[1];
  const float* v     = (const float*)d_in[2];
  const float* mask  = (const float*)d_in[3];
  const float* noise = (const float*)d_in[4];
  const float* wq    = (const float*)d_in[5];
  const float* bq    = (const float*)d_in[6];
  const float* wk    = (const float*)d_in[7];
  const float* bk    = (const float*)d_in[8];
  const float* wv    = (const float*)d_in[9];
  const float* bv    = (const float*)d_in[10];
  const float* wo    = (const float*)d_in[11];
  const float* bo    = (const float*)d_in[12];

  float* outp  = (float*)d_out;                              // [B,S,D]
  float* attnp = outp + (size_t)BATCH * SEQ * DMODEL;        // [B,H,S,S]

  // Workspace layout (256 MB total):
  //   qbf: bf16 [B,H,S,DEPTH] (64 MB)  -- reused as out_head after attention
  //   kbf: bf16 [B,H,S,DEPTH] (64 MB)
  //   vfp: f32  [B,H,S,DEPTH] (128 MB)
  char* ws = (char*)d_ws;
  bf16_t* qbf = (bf16_t*)ws;
  bf16_t* kbf = (bf16_t*)(ws + (size_t)67108864);
  float*  vfp = (float*)(ws + (size_t)134217728);

  const dim3 blk(256);
  const dim3 grid_proj(DMODEL / 128, (BATCH * SEQ) / 128);   // (8, 256)

  proj_kernel<0><<<grid_proj, blk, 0, stream>>>(q, wq, bq, (void*)qbf);
  proj_kernel<0><<<grid_proj, blk, 0, stream>>>(k, wk, bk, (void*)kbf);
  proj_kernel<1><<<grid_proj, blk, 0, stream>>>(v, wv, bv, (void*)vfp);

  const size_t attn_lds = (size_t)4 * 128 * 136 * sizeof(bf16_t);  // 139264 B
  attn_kernel<<<BATCH * NHEADS, blk, attn_lds, stream>>>(
      qbf, kbf, vfp, noise, mask, attnp, /*out_head=*/qbf);

  final_proj_kernel<<<grid_proj, blk, 0, stream>>>(qbf, wo, bo, outp);
}
